// SSAAttention_43705587204682
// MI455X (gfx1250) — matrix operand order repaired and checked by reading, not compiled, not run on hardware
//
#include <hip/hip_runtime.h>
#include <hip/hip_bf16.h>

// ---------------- problem constants ----------------
#define BB    2
#define SS    2048
#define HIDC  2048
#define NH    16
#define NKV   8
#define DH    128
#define MT    (BB*SS)        // 4096 tokens
#define KDIM  (NKV*DH)       // 1024
#define WIN   128
#define KCNT  675            // int(0.33*2048)
#define NEGV  (-10000.0f)
#define BROWS 72             // padded LDS row stride in halves (144 B)

typedef __attribute__((ext_vector_type(16))) __bf16 v16bf;
typedef __attribute__((ext_vector_type(8)))  float  v8f;

union frag16 { unsigned u[8]; v16bf v; };

static __device__ __forceinline__ v8f wmma_bf16(const frag16& a, const frag16& b, v8f c) {
  return __builtin_amdgcn_wmma_f32_16x16x32_bf16(false, a.v, false, b.v, (short)0, c, false, false);
}

static __device__ __forceinline__ unsigned short f2bf(float f) {
  unsigned u = __builtin_bit_cast(unsigned, f);
  u += 0x7FFFu + ((u >> 16) & 1u);          // round-to-nearest-even
  return (unsigned short)(u >> 16);
}
static __device__ __forceinline__ v8f vzero() {
  v8f z;
  #pragma unroll
  for (int i = 0; i < 8; ++i) z[i] = 0.0f;
  return z;
}
// flat LDS pointer -> 32-bit LDS byte offset (aperture maps addr[31:0] to LDS)
static __device__ __forceinline__ unsigned lds_off32(const void* p) {
  return (unsigned)(unsigned long long)p;
}
static __device__ __forceinline__ void async_ld_b128(unsigned lds_byte_off, const void* gsrc) {
  asm volatile("global_load_async_to_lds_b128 %0, %1, off"
               :: "v"(lds_byte_off), "v"(gsrc) : "memory");
}
static __device__ __forceinline__ void wait_async0() {
  asm volatile("s_wait_asynccnt 0x0" ::: "memory");
}

// ---------------- conversion / transpose kernels ----------------
__global__ void k_cvt_bf16(const float* __restrict__ in, unsigned short* __restrict__ out, int n) {
  int i = blockIdx.x * blockDim.x + threadIdx.x;
  if (i < n) out[i] = f2bf(in[i]);
}

// W [K][N] f32  ->  Wt [N][K] bf16   (output-coalesced in k)
__global__ void k_transpose_w(const float* __restrict__ W, unsigned short* __restrict__ Wt,
                              int K, int N) {
  int i = blockIdx.x * blockDim.x + threadIdx.x;
  if (i >= K * N) return;
  int n = i / K, k = i - n * K;
  Wt[i] = f2bf(W[(size_t)k * N + n]);
}

// V [token][kv*D+d] bf16 -> Vt [b][kv][d][s] bf16
__global__ void k_transpose_v(const unsigned short* __restrict__ V, unsigned short* __restrict__ Vt) {
  int i = blockIdx.x * blockDim.x + threadIdx.x;
  if (i >= BB * NKV * DH * SS) return;
  int s  = i & (SS - 1);
  int d  = (i >> 11) & (DH - 1);
  int kv = (i >> 18) & (NKV - 1);
  int b  = i >> 21;
  Vt[i] = V[((size_t)(b * SS + s)) * KDIM + kv * DH + d];
}

// ---------------- TLS mask pipeline ----------------
__global__ void k_mean(const float* __restrict__ X, float* __restrict__ mean) {
  int i = blockIdx.x * blockDim.x + threadIdx.x;     // b*HID + c
  if (i >= BB * HIDC) return;
  int b = i / HIDC, c = i - b * HIDC;
  float s = 0.f;
  const float* p = X + (size_t)b * SS * HIDC + c;
  for (int t = 0; t < SS; ++t) s += p[(size_t)t * HIDC];
  mean[i] = s * (1.0f / SS);
}

__global__ void k_mnorm(const float* __restrict__ mean, float* __restrict__ mnorm) {
  int b = blockIdx.x;
  __shared__ float red[256];
  float s = 0.f;
  for (int c = threadIdx.x; c < HIDC; c += 256) {
    float v = mean[b * HIDC + c];
    s += v * v;
  }
  red[threadIdx.x] = s; __syncthreads();
  for (int o = 128; o > 0; o >>= 1) {
    if (threadIdx.x < o) red[threadIdx.x] += red[threadIdx.x + o];
    __syncthreads();
  }
  if (threadIdx.x == 0) mnorm[b] = sqrtf(red[0]);
}

// one wave per token: persistence + diversity score
__global__ __launch_bounds__(256) void k_score(const float* __restrict__ X,
                                               const float* __restrict__ mean,
                                               const float* __restrict__ mnorm,
                                               float* __restrict__ scores) {
  int lane = threadIdx.x & 31;
  int gw = blockIdx.x * 8 + (threadIdx.x >> 5);   // b*S + s
  int b = gw / SS, s = gw - b * SS;
  const float* row = X + ((size_t)b * SS + s) * HIDC;
  const float* mrow = mean + b * HIDC;
  float sq = 0.f, dt = 0.f;
  for (int c = lane; c < HIDC; c += 32) {
    float x = row[c];
    sq += x * x;
    dt += x * mrow[c];
  }
  #pragma unroll
  for (int m = 16; m > 0; m >>= 1) {
    sq += __shfl_xor(sq, m, 32);
    dt += __shfl_xor(dt, m, 32);
  }
  if (lane == 0) {
    float nrm = sqrtf(sq);
    float div = -dt / (nrm * mnorm[b] + 1e-6f);
    scores[gw] = 0.7f * nrm + 0.3f * div;
  }
}

// per batch: bitonic sort in LDS -> threshold -> landmark flags -> dilate(2) -> tile flags
__global__ __launch_bounds__(1024) void k_topk(const float* __restrict__ scores,
                                               unsigned char* __restrict__ lmd,
                                               unsigned* __restrict__ lmtile) {
  int b = blockIdx.x, t = threadIdx.x;
  __shared__ float sm[SS];
  __shared__ unsigned char fl[SS];
  __shared__ unsigned char dl[SS];
  sm[t]        = scores[b * SS + t];
  sm[t + 1024] = scores[b * SS + t + 1024];
  __syncthreads();
  for (int k = 2; k <= SS; k <<= 1) {
    for (int j = k >> 1; j > 0; j >>= 1) {
      #pragma unroll
      for (int r = 0; r < 2; ++r) {
        int e = t + r * 1024;
        int p = e ^ j;
        if (p > e) {
          bool up = ((e & k) == 0);        // ascending sort
          float a = sm[e], c = sm[p];
          bool swap = up ? (a > c) : (a < c);
          if (swap) { sm[e] = c; sm[p] = a; }
        }
      }
      __syncthreads();
    }
  }
  float thr = sm[SS - KCNT];               // k-th largest
  #pragma unroll
  for (int r = 0; r < 2; ++r) {
    int s = t + r * 1024;
    fl[s] = (scores[b * SS + s] >= thr) ? 1 : 0;
  }
  __syncthreads();
  #pragma unroll
  for (int r = 0; r < 2; ++r) {
    int s = t + r * 1024;
    unsigned char v = 0;
    for (int o = -2; o <= 2; ++o) {        // 2 hops of +/-1
      int ss2 = s + o;
      if (ss2 >= 0 && ss2 < SS) v |= fl[ss2];
    }
    dl[s] = v;
    lmd[b * SS + s] = v;
  }
  __syncthreads();
  if (t < SS / 32) {
    unsigned any = 0;
    for (int i = 0; i < 32; ++i) any |= dl[t * 32 + i];
    lmtile[b * (SS / 32) + t] = any;
  }
}

// ---------------- bf16 WMMA GEMM: C[M,N] = A[M,K] * Bt[N,K]^T ----------------
// Block = 128(M) x 64(N): 8 waves, each a 16x64 strip. B tile (64n x 32k) is
// double-buffered in LDS via global_load_async_to_lds_b128 (ASYNCcnt).
static __device__ __forceinline__ void gemm_tile_step(
    int i, int steps, const unsigned short* curbuf,
    frag16& afc, frag16& afn, unsigned nxt_lds,
    const unsigned short* gsrc_next, const unsigned short* Arow,
    int kh, int nlo, v8f* acc)
{
  bool more = (i + 1) < steps;
  if (more) {
    async_ld_b128(nxt_lds, gsrc_next);           // B tile i+1 -> other LDS buffer
    int k0n = (i + 1) << 5;
    #pragma unroll
    for (int v = 0; v < 8; ++v) {                // A fragment i+1 -> other regs
      int kk = k0n + ((v >> 2) << 4) + (kh << 3) + ((v & 3) << 1);
      afn.u[v] = *(const unsigned*)(Arow + kk);
    }
  }
  #pragma unroll
  for (int t = 0; t < 4; ++t) {
    frag16 bfr;
    const unsigned short* brow = curbuf + (t * 16 + nlo) * BROWS + (kh << 3);
    #pragma unroll
    for (int v = 0; v < 8; ++v)
      bfr.u[v] = *(const unsigned*)(brow + ((v >> 2) << 4) + ((v & 3) << 1));
    acc[t] = wmma_bf16(afc, bfr, acc[t]);
  }
  if (more) wait_async0();
  __syncthreads();
}

__global__ __launch_bounds__(256) void k_gemm_bf16(const unsigned short* __restrict__ A,
                                                   const unsigned short* __restrict__ Bt,
                                                   unsigned short* __restrict__ Cb,
                                                   float* __restrict__ Cf,
                                                   int M, int N, int K, int store_f32) {
  __shared__ unsigned short Bs[2][64 * BROWS];
  int tid  = threadIdx.x;
  int lane = tid & 31, wv = tid >> 5;
  int nb = N >> 6;
  int bm = blockIdx.x / nb;
  int n0 = (blockIdx.x - bm * nb) << 6;
  int m0 = (bm << 7) + (wv << 4);
  int nlo = lane & 15, kh = lane >> 4;

  // per-thread async-copy assignment: row cn (0..63), 16-byte chunk cc (0..3)
  int cn = tid >> 2, cc = tid & 3;
  const unsigned short* gsrc = Bt + (size_t)(n0 + cn) * K + cc * 8;
  unsigned ldst0 = lds_off32(&Bs[0][0]) + (unsigned)(cn * (BROWS * 2) + cc * 16);
  unsigned ldst1 = ldst0 + (unsigned)(64 * BROWS * 2);

  v8f acc[4];
  #pragma unroll
  for (int t = 0; t < 4; ++t) acc[t] = vzero();

  const unsigned short* Arow = A + (size_t)(m0 + nlo) * K;

  // prologue: tile 0 -> buf0; A fragment 0
  async_ld_b128(ldst0, gsrc);
  frag16 af0, af1;
  #pragma unroll
  for (int v = 0; v < 8; ++v) {
    int kk = ((v >> 2) << 4) + (kh << 3) + ((v & 3) << 1);
    af0.u[v] = *(const unsigned*)(Arow + kk);
  }
  wait_async0();
  __syncthreads();

  int steps = K >> 5;                       // even (K=2048 -> 64)
  for (int i = 0; i < steps; i += 2) {
    gemm_tile_step(i,     steps, &Bs[0][0], af0, af1, ldst1,
                   gsrc + (size_t)(i + 1) * 32, Arow, kh, nlo, acc);
    gemm_tile_step(i + 1, steps, &Bs[1][0], af1, af0, ldst0,
                   gsrc + (size_t)(i + 2) * 32, Arow, kh, nlo, acc);
  }

  #pragma unroll
  for (int t = 0; t < 4; ++t)
    #pragma unroll
    for (int v = 0; v < 8; ++v) {
      int m = m0 + (kh << 3) + v;
      int n = n0 + t * 16 + nlo;
      if (store_f32) Cf[(size_t)m * N + n] = acc[t][v];
      else           Cb[(size_t)m * N + n] = f2bf(acc[t][v]);
    }
}

// ---------------- flash attention with TLS sparsity ----------------
__global__ __launch_bounds__(256) void k_attn(const unsigned short* __restrict__ Qb,
                                              const unsigned short* __restrict__ Kb,
                                              const unsigned short* __restrict__ Vt,
                                              const unsigned char* __restrict__ lmd,
                                              const unsigned* __restrict__ lmtile,
                                              unsigned short* __restrict__ Ab) {
  __shared__ unsigned short Pbuf[8][16 * 32];
  int lane = threadIdx.x & 31;
  int wv   = threadIdx.x >> 5;
  int gw   = blockIdx.x * 8 + wv;           // b*H*(S/16) waves
  int qt = gw & 127;
  int h  = (gw >> 7) & 15;
  int b  = gw >> 11;
  int kvh = h >> 1;                          // GQA: H/KVH = 2
  int q0 = qt << 4;
  int nlo = lane & 15, kh = lane >> 4;
  int rbase = kh << 3;
  const float scale = 0.08838834764831845f;  // 1/sqrt(128)

  // Q A-fragments for the 4 k-chunks of D=128
  frag16 qf[4];
  const unsigned short* Qrow = Qb + ((size_t)(b * SS) + q0 + nlo) * HIDC + h * DH;
  #pragma unroll
  for (int c = 0; c < 4; ++c)
    #pragma unroll
    for (int v = 0; v < 8; ++v) {
      int d = c * 32 + ((v >> 2) << 4) + (kh << 3) + ((v & 3) << 1);
      qf[c].u[v] = *(const unsigned*)(Qrow + d);
    }

  v8f oacc[8];
  #pragma unroll
  for (int dc = 0; dc < 8; ++dc) oacc[dc] = vzero();
  float rmax[8], rsum[8];
  #pragma unroll
  for (int v = 0; v < 8; ++v) { rmax[v] = -1e30f; rsum[v] = 0.f; }

  int ktmax = (q0 + 15) >> 5;
  for (int kt = 0; kt <= ktmax; ++kt) {
    int k0 = kt << 5;
    bool local_ov = (k0 + 31) >= (q0 - (WIN - 1));
    bool lmrel    = lmtile[b * (SS / 32) + kt] != 0;
    if (!(local_ov || lmrel)) continue;     // wave-uniform skip

    // ---- scores: 2 halves x 4 d-chunks of QK^T WMMA ----
    v8f sacc[2];
    sacc[0] = vzero(); sacc[1] = vzero();
    #pragma unroll
    for (int half = 0; half < 2; ++half) {
      const unsigned short* Krow =
          Kb + ((size_t)(b * SS) + k0 + half * 16 + nlo) * KDIM + kvh * DH;
      #pragma unroll
      for (int c = 0; c < 4; ++c) {
        frag16 kf;
        #pragma unroll
        for (int v = 0; v < 8; ++v)
          kf.u[v] = *(const unsigned*)(Krow + c * 32 + ((v >> 2) << 4) + (kh << 3) + ((v & 3) << 1));
        sacc[half] = wmma_bf16(qf[c], kf, sacc[half]);
      }
    }

    // ---- mask + online softmax (rows live across 16-lane groups) ----
    bool lmk0 = lmd[b * SS + k0 + nlo] != 0;
    bool lmk1 = lmd[b * SS + k0 + 16 + nlo] != 0;
    #pragma unroll
    for (int v = 0; v < 8; ++v) {
      int q = q0 + rbase + v;
      int key0 = k0 + nlo, key1 = k0 + 16 + nlo;
      bool a0 = (key0 <= q) && (((q - key0) < WIN) || lmk0);
      bool a1 = (key1 <= q) && (((q - key1) < WIN) || lmk1);
      float s0 = sacc[0][v] * scale + (a0 ? 0.f : NEGV);
      float s1 = sacc[1][v] * scale + (a1 ? 0.f : NEGV);
      float m = fmaxf(s0, s1);
      #pragma unroll
      for (int o = 8; o > 0; o >>= 1) m = fmaxf(m, __shfl_xor(m, o, 16));
      float nm   = fmaxf(rmax[v], m);
      float corr = __expf(rmax[v] - nm);
      rmax[v] = nm;
      float p0 = __expf(s0 - nm);
      float p1 = __expf(s1 - nm);
      float rs = p0 + p1;
      #pragma unroll
      for (int o = 8; o > 0; o >>= 1) rs += __shfl_xor(rs, o, 16);
      rsum[v] = rsum[v] * corr + rs;
      #pragma unroll
      for (int dc = 0; dc < 8; ++dc) oacc[dc][v] *= corr;
      int row = rbase + v;
      Pbuf[wv][row * 32 + nlo]      = f2bf(p0);
      Pbuf[wv][row * 32 + 16 + nlo] = f2bf(p1);
    }
    asm volatile("s_wait_dscnt 0" ::: "memory");

    // ---- P A-fragment from LDS ----
    frag16 pf;
    #pragma unroll
    for (int v = 0; v < 8; ++v) {
      int kk = ((v >> 2) << 4) + (kh << 3) + ((v & 3) << 1);
      pf.u[v] = *(const unsigned*)(&Pbuf[wv][nlo * 32 + kk]);
    }

    // ---- PV: 8 d-chunks of 16 ----
    #pragma unroll
    for (int dc = 0; dc < 8; ++dc) {
      const unsigned short* Vrow =
          Vt + (((size_t)(b * NKV + kvh) * DH) + dc * 16 + nlo) * SS + k0;
      frag16 vf;
      #pragma unroll
      for (int v = 0; v < 8; ++v)
        vf.u[v] = *(const unsigned*)(Vrow + ((v >> 2) << 4) + (kh << 3) + ((v & 3) << 1));
      oacc[dc] = wmma_bf16(pf, vf, oacc[dc]);
    }
  }

  // ---- normalize + store bf16 ----
  #pragma unroll
  for (int dc = 0; dc < 8; ++dc)
    #pragma unroll
    for (int v = 0; v < 8; ++v) {
      float val = oacc[dc][v] / rsum[v];
      int q = q0 + rbase + v;
      Ab[((size_t)(b * SS) + q) * HIDC + h * DH + dc * 16 + nlo] = f2bf(val);
    }
}

// ---------------- host orchestration ----------------
extern "C" void kernel_launch(void* const* d_in, const int* in_sizes, int n_in,
                              void* d_out, int out_size, void* d_ws, size_t ws_size,
                              hipStream_t stream) {
  const float* X  = (const float*)d_in[0];
  const float* Wq = (const float*)d_in[1];
  const float* Wk = (const float*)d_in[2];
  const float* Wv = (const float*)d_in[3];
  const float* Wo = (const float*)d_in[4];
  float* out = (float*)d_out;

  char* base = (char*)d_ws;
  size_t off = 0;
  auto alloc = [&](size_t bytes) -> char* {
    char* p = base + off;
    off = (off + bytes + 255) & ~(size_t)255;
    return p;
  };
  unsigned short* Xb   = (unsigned short*)alloc((size_t)MT * HIDC * 2);
  unsigned short* Wqt  = (unsigned short*)alloc((size_t)HIDC * HIDC * 2);
  unsigned short* Wkt  = (unsigned short*)alloc((size_t)KDIM * HIDC * 2);
  unsigned short* Wvt  = (unsigned short*)alloc((size_t)KDIM * HIDC * 2);
  unsigned short* Wot  = (unsigned short*)alloc((size_t)HIDC * HIDC * 2);
  unsigned short* Qb   = (unsigned short*)alloc((size_t)MT * HIDC * 2);
  unsigned short* Kb   = (unsigned short*)alloc((size_t)MT * KDIM * 2);
  unsigned short* Vb   = (unsigned short*)alloc((size_t)MT * KDIM * 2);
  unsigned short* Vtb  = (unsigned short*)alloc((size_t)MT * KDIM * 2);
  unsigned short* Abuf = (unsigned short*)alloc((size_t)MT * HIDC * 2);
  float*         meanb = (float*)alloc((size_t)BB * HIDC * 4);
  float*         mnb   = (float*)alloc((size_t)BB * 4);
  float*         scb   = (float*)alloc((size_t)BB * SS * 4);
  unsigned char* lmdb  = (unsigned char*)alloc((size_t)BB * SS);
  unsigned*      lmtb  = (unsigned*)alloc((size_t)BB * (SS / 32) * 4);

  // 1. conversions
  k_cvt_bf16<<<(MT * HIDC) / 256, 256, 0, stream>>>(X, Xb, MT * HIDC);
  k_transpose_w<<<(HIDC * HIDC) / 256, 256, 0, stream>>>(Wq, Wqt, HIDC, HIDC);
  k_transpose_w<<<(HIDC * KDIM) / 256, 256, 0, stream>>>(Wk, Wkt, HIDC, KDIM);
  k_transpose_w<<<(HIDC * KDIM) / 256, 256, 0, stream>>>(Wv, Wvt, HIDC, KDIM);
  k_transpose_w<<<(HIDC * HIDC) / 256, 256, 0, stream>>>(Wo, Wot, HIDC, HIDC);

  // 2. TLS mask pipeline
  k_mean<<<(BB * HIDC) / 256, 256, 0, stream>>>(X, meanb);
  k_mnorm<<<BB, 256, 0, stream>>>(meanb, mnb);
  k_score<<<(BB * SS) / 8, 256, 0, stream>>>(X, meanb, mnb, scb);
  k_topk<<<BB, 1024, 0, stream>>>(scb, lmdb, lmtb);

  // 3. projections (WMMA bf16, async-LDS double-buffered B tiles)
  k_gemm_bf16<<<(MT / 128) * (HIDC / 64), 256, 0, stream>>>(Xb, Wqt, Qb, nullptr, MT, HIDC, HIDC, 0);
  k_gemm_bf16<<<(MT / 128) * (KDIM / 64), 256, 0, stream>>>(Xb, Wkt, Kb, nullptr, MT, KDIM, HIDC, 0);
  k_gemm_bf16<<<(MT / 128) * (KDIM / 64), 256, 0, stream>>>(Xb, Wvt, Vb, nullptr, MT, KDIM, HIDC, 0);
  k_transpose_v<<<(BB * NKV * DH * SS) / 256, 256, 0, stream>>>(Vb, Vtb);

  // 4. sparse flash attention (WMMA bf16)
  k_attn<<<(BB * NH * (SS / 16)) / 8, 256, 0, stream>>>(Qb, Kb, Vtb, lmdb, lmtb, Abuf);

  // 5. output projection -> f32 d_out
  k_gemm_bf16<<<(MT / 128) * (HIDC / 64), 256, 0, stream>>>(Abuf, Wot, nullptr, out, MT, HIDC, HIDC, 1);
}
